// RobustFill_4990751998516
// MI455X (gfx1250) — matrix operand order repaired and checked by reading, not compiled, not run on hardware
//
#include <hip/hip_runtime.h>
#include <hip/hip_bf16.h>
#include <stdint.h>

// Problem dimensions (match reference)
#define D_V    128
#define D_EMB  128
#define D_H    1024
#define D_P    1024
#define D_PLEN 16
#define D_B    128
#define D_E    4
#define D_F    (D_B * D_E)      // 512
#define D_L    128
#define D_G4H  (4 * D_H)        // 4096
#define D_KCAT (D_EMB + D_H)    // 1152 : [x | h] concatenated K

typedef __attribute__((ext_vector_type(16))) __bf16 v16bf;
typedef __attribute__((ext_vector_type(8)))  float  v8f;

__device__ __forceinline__ float sigmoidf_(float x) {
    return 1.0f / (1.0f + __expf(-x));
}

// ---------------------------------------------------------------------------
// WMMA GEMM:  C[M,N] = A[M,K](bf16) * Bp[K,N](bf16, pre-packed)  (+ bias[N])
//
// Bp layout: tile (kc, nt) of shape 32(K) x 16(N); per lane (contiguous K, this code's order, not the ISA table order):
//   N = lane&15, K = kc*32 + (lane>=16 ? 16:0) + e, e = 0..15 contiguous bf16.
// For a fixed kc, 4 consecutive nt tiles are one CONTIGUOUS 4KB panel, which
// we double-buffer through LDS with GLOBAL_LOAD_ASYNC_TO_LDS_B128 (ASYNCcnt),
// shared by all 8 waves of the block (8x B reuse vs per-wave global loads).
//
// Block = 256 threads = 8 waves; block tile = 128(M) x 64(N); wave w owns
// M-tile (bm*8 + w). Requires M%128==0, N%64==0, K%32==0.
// ---------------------------------------------------------------------------
__global__ __launch_bounds__(256) void wmma_gemm_kernel(
    const __bf16* __restrict__ A, int lda,
    const __bf16* __restrict__ Bp,
    const float*  __restrict__ bias,
    float* __restrict__ C, int ldc,
    int M, int N, int K)
{
    __shared__ __bf16 smem[2][2048];   // 2 x 4KB: double-buffered 32x64 panel

    const int tid  = threadIdx.x;
    const int lane = tid & 31;
    const int wave = tid >> 5;
    const int ngroups = N >> 6;                 // 64-wide N groups
    const int bm = blockIdx.x / ngroups;
    const int ng = blockIdx.x - bm * ngroups;
    const int mt = bm * 8 + wave;

    const int ln15 = lane & 15;
    const int hi   = lane >> 4;                 // 0/1: K-half selector
    const int row  = mt * 16 + ln15;
    const int kchunks = K >> 5;
    const int ntiles  = N >> 4;

    // B panel source for chunk kc: Bp + (kc*ntiles + ng*4)*512, 2048 bf16.
    const __bf16* bsrc0 = Bp + ((size_t)ng * 4) * 512 + (size_t)tid * 8;
    const size_t  bstep = (size_t)ntiles * 512;        // per-kc advance

    v8f acc[4] = {};
    union AU { unsigned int u[8]; v16bf v; };
    union BU { uint4 q[2];        v16bf v; };

    // prefetch kc = 0 (async copy to LDS, 16B per thread)
    {
        unsigned dst = (unsigned)(uintptr_t)(&smem[0][tid * 8]);
        asm volatile("global_load_async_to_lds_b128 %0, %1, off"
                     :: "v"(dst), "v"(bsrc0) : "memory");
    }

    for (int kc = 0; kc < kchunks; ++kc) {
        asm volatile("s_wait_asynccnt 0x0" ::: "memory");
        __syncthreads();                        // staged panel visible to all
        if (kc + 1 < kchunks) {                 // prefetch next panel (other buf)
            unsigned dst =
                (unsigned)(uintptr_t)(&smem[(kc + 1) & 1][tid * 8]);
            const __bf16* src = bsrc0 + (size_t)(kc + 1) * bstep;
            asm volatile("global_load_async_to_lds_b128 %0, %1, off"
                         :: "v"(dst), "v"(src) : "memory");
        }

        // ---- A tile 16x32 bf16 (contiguous K: K = 16*hi + e, matches packed B)
        AU a;
        const __bf16* arow = A + (size_t)row * lda + (kc << 5) + (hi << 4);
        #pragma unroll
        for (int v = 0; v < 8; ++v) {
            const int koff = v << 1;
            a.u[v] = *(const unsigned int*)(arow + koff);
        }

        // ---- 4 B subtiles from LDS (2x ds_load_b128 each) + WMMA
        const __bf16* bbase = &smem[kc & 1][0];
        #pragma unroll
        for (int sub = 0; sub < 4; ++sub) {
            const __bf16* bptr = bbase + (sub * 32 + lane) * 16;
            BU b;
            b.q[0] = *(const uint4*)(bptr);
            b.q[1] = *(const uint4*)(bptr + 8);
            acc[sub] = __builtin_amdgcn_wmma_f32_16x16x32_bf16(
                false, a.v, false, b.v, (short)0, acc[sub], false, false);
        }
        // No trailing barrier needed: each wave's ds_loads are consumed by
        // WMMA (s_wait_dscnt) before it reaches the next-iteration barrier.
    }

    // ---- store C (f32 16x16 layout: VGPR v -> M = v + 8*hi, N = ln15)
    #pragma unroll
    for (int sub = 0; sub < 4; ++sub) {
        const int col  = (ng << 6) + (sub << 4) + ln15;
        const float bb = bias ? bias[col] : 0.0f;
        const int rbase = mt * 16 + (hi << 3);
        #pragma unroll
        for (int v = 0; v < 8; ++v) {
            C[(size_t)(rbase + v) * ldc + col] = acc[sub][v] + bb;
        }
    }
}

// ---------------------------------------------------------------------------
// Pack B = [Wa ; Wb]^T into WMMA-native tiled bf16 layout.
// Wa: [N, Ka] f32 row-major, Wb: [N, Kb] f32 row-major (either may be absent
// via Ka==0).  dst element order matches wmma_gemm_kernel's B fetch exactly.
// ---------------------------------------------------------------------------
__global__ __launch_bounds__(256) void pack_b_kernel(
    const float* __restrict__ Wa, int Ka,
    const float* __restrict__ Wb, int Kb,
    int N, __bf16* __restrict__ dst)
{
    const int tid  = blockIdx.x * blockDim.x + threadIdx.x;
    const int Ktot = Ka + Kb;
    if (tid >= Ktot * N) return;
    const int e    = tid & 15;
    const int lane = (tid >> 4) & 31;
    const int tile = tid >> 9;
    const int ntiles = N >> 4;
    const int kt = tile / ntiles;
    const int nt = tile - kt * ntiles;
    const int k  = (kt << 5) + ((lane >> 4) << 4) + e;
    const int j  = (nt << 4) + (lane & 15);
    const float val = (k < Ka) ? Wa[(size_t)j * Ka + k]
                               : Wb[(size_t)j * Kb + (k - Ka)];
    dst[tid] = (__bf16)val;
}

__global__ void f32_to_bf16_kernel(const float* __restrict__ src,
                                   __bf16* __restrict__ dst, int n)
{
    int i = blockIdx.x * blockDim.x + threadIdx.x;
    if (i < n) dst[i] = (__bf16)src[i];
}

__global__ void zero_u32_kernel(unsigned int* __restrict__ p, int n)
{
    int i = blockIdx.x * blockDim.x + threadIdx.x;
    if (i < n) p[i] = 0u;
}

// Gather embedded tokens of timestep t into the x-part of the A buffer.
__global__ void gather_x_kernel(const int* __restrict__ tokens, int t,
                                const __bf16* __restrict__ emb,
                                __bf16* __restrict__ Abuf)
{
    int tid = blockIdx.x * blockDim.x + threadIdx.x;   // D_F * D_EMB
    int f = tid >> 7;
    int k = tid & 127;
    int tok = tokens[f * D_L + t];
    Abuf[(size_t)f * D_KCAT + k] = emb[tok * D_EMB + k];
}

// Encoder LSTM gates with pack_sequence masking; writes h (bf16) into Abuf.
__global__ void lstm_gates_enc_kernel(const float* __restrict__ g,
                                      const float* __restrict__ b,
                                      const int*   __restrict__ lengths, int t,
                                      float*  __restrict__ c,
                                      __bf16* __restrict__ Abuf)
{
    int tid = blockIdx.x * blockDim.x + threadIdx.x;   // D_F * D_H
    int f = tid >> 10;
    int j = tid & 1023;
    if (t >= lengths[f]) return;                       // masked: keep h, c
    const float* gr = g + (size_t)f * D_G4H;
    float gi = gr[j]           + b[j];
    float gf = gr[D_H + j]     + b[D_H + j];
    float gg = gr[2 * D_H + j] + b[2 * D_H + j];
    float go = gr[3 * D_H + j] + b[3 * D_H + j];
    float cn = sigmoidf_(gf) * c[tid] + sigmoidf_(gi) * tanhf(gg);
    float hn = sigmoidf_(go) * tanhf(cn);
    c[tid] = cn;
    Abuf[(size_t)f * D_KCAT + D_EMB + j] = (__bf16)hn;
}

// Decoder gates: g_total = g + pproj (pproj = prev_h@W_ih_p^T + b_p). No mask.
__global__ void lstm_gates_dec_kernel(const float* __restrict__ g,
                                      const float* __restrict__ pproj,
                                      float*  __restrict__ c,
                                      __bf16* __restrict__ hdec)
{
    int tid = blockIdx.x * blockDim.x + threadIdx.x;   // D_F * D_H
    int f = tid >> 10;
    int j = tid & 1023;
    const float* gr = g     + (size_t)f * D_G4H;
    const float* pr = pproj + (size_t)f * D_G4H;
    float gi = gr[j]           + pr[j];
    float gf = gr[D_H + j]     + pr[D_H + j];
    float gg = gr[2 * D_H + j] + pr[2 * D_H + j];
    float go = gr[3 * D_H + j] + pr[3 * D_H + j];
    float cn = sigmoidf_(gf) * c[tid] + sigmoidf_(gi) * tanhf(gg);
    float hn = sigmoidf_(go) * tanhf(cn);
    c[tid] = cn;
    hdec[tid] = (__bf16)hn;
}

__global__ void copy_h_kernel(const __bf16* __restrict__ Abuf,
                              __bf16* __restrict__ hdec)
{
    int tid = blockIdx.x * blockDim.x + threadIdx.x;   // D_F * D_H
    int f = tid >> 10;
    int j = tid & 1023;
    hdec[tid] = Abuf[(size_t)f * D_KCAT + D_EMB + j];
}

// pooled[b, j] = max_e tanh(u[b*E+e, j] + b_mp[j])  (bf16 out for next GEMM)
__global__ void pool_kernel(const float* __restrict__ u,
                            const float* __restrict__ bmp,
                            __bf16* __restrict__ pooled)
{
    int tid = blockIdx.x * blockDim.x + threadIdx.x;   // D_B * D_H
    int bidx = tid >> 10;
    int j = tid & 1023;
    float bj = bmp[j];
    float m = -1e30f;
    #pragma unroll
    for (int e = 0; e < D_E; ++e) {
        float v = tanhf(u[(size_t)(bidx * D_E + e) * D_H + j] + bj);
        m = fmaxf(m, v);
    }
    pooled[tid] = (__bf16)m;
}

// ---------------------------------------------------------------------------
extern "C" void kernel_launch(void* const* d_in, const int* in_sizes, int n_in,
                              void* d_out, int out_size, void* d_ws, size_t ws_size,
                              hipStream_t stream)
{
    (void)in_sizes; (void)n_in; (void)out_size; (void)ws_size;

    const int*   tokens_in   = (const int*)  d_in[0];
    const int*   lengths_in  = (const int*)  d_in[1];
    const int*   tokens_out  = (const int*)  d_in[2];
    const int*   lengths_out = (const int*)  d_in[3];
    const float* embedding   = (const float*)d_in[4];
    const float* W_ih_in     = (const float*)d_in[5];
    const float* W_hh_in     = (const float*)d_in[6];
    const float* b_in        = (const float*)d_in[7];
    const float* W_ih_out    = (const float*)d_in[8];
    const float* W_hh_out    = (const float*)d_in[9];
    const float* b_out       = (const float*)d_in[10];
    const float* W_ih_p      = (const float*)d_in[11];
    const float* W_hh_p      = (const float*)d_in[12];
    const float* b_p         = (const float*)d_in[13];
    const float* W_mp        = (const float*)d_in[14];
    const float* b_mp        = (const float*)d_in[15];
    const float* W_sm        = (const float*)d_in[16];
    const float* b_sm        = (const float*)d_in[17];
    float* out = (float*)d_out;

    // --- workspace carve-up (256B aligned) ---
    char* wp = (char*)d_ws;
    auto alloc = [&](size_t bytes) -> void* {
        void* r = (void*)wp;
        wp += (bytes + 255) & ~(size_t)255;
        return r;
    };
    __bf16* Bcat_in  = (__bf16*)alloc((size_t)D_KCAT * D_G4H * 2); // [Wih|Whh]^T
    __bf16* Bcat_out = (__bf16*)alloc((size_t)D_KCAT * D_G4H * 2);
    __bf16* Bp_ihp   = (__bf16*)alloc((size_t)D_H * D_G4H * 2);
    __bf16* Bp_hhp   = (__bf16*)alloc((size_t)D_H * D_G4H * 2);
    __bf16* Bp_mp    = (__bf16*)alloc((size_t)D_H * D_H * 2);
    __bf16* Bp_sm    = (__bf16*)alloc((size_t)D_H * D_P * 2);
    __bf16* emb_bf   = (__bf16*)alloc((size_t)D_V * D_EMB * 2);
    __bf16* Abuf     = (__bf16*)alloc((size_t)D_F * D_KCAT * 2);   // [x | h]
    float*  gbuf     = (float*) alloc((size_t)D_F * D_G4H * 4);
    float*  pproj    = (float*) alloc((size_t)D_F * D_G4H * 4);
    float*  c_f32    = (float*) alloc((size_t)D_F * D_H * 4);
    __bf16* hdec     = (__bf16*)alloc((size_t)D_F * D_H * 2);
    float*  ubuf     = (float*) alloc((size_t)D_F * D_H * 4);
    __bf16* pooled   = (__bf16*)alloc((size_t)D_B * D_H * 2);

    const int TB = 256;
    #define GRID(n) (((n) + TB - 1) / TB)

    // --- init state (deterministic per call) ---
    zero_u32_kernel<<<GRID(D_F * D_H), TB, 0, stream>>>((unsigned int*)c_f32, D_F * D_H);
    zero_u32_kernel<<<GRID(D_F * D_KCAT / 2), TB, 0, stream>>>((unsigned int*)Abuf, D_F * D_KCAT / 2);

    // --- one-time weight packing (bf16, WMMA-native tiled layout) ---
    f32_to_bf16_kernel<<<GRID(D_V * D_EMB), TB, 0, stream>>>(embedding, emb_bf, D_V * D_EMB);
    pack_b_kernel<<<GRID(D_KCAT * D_G4H), TB, 0, stream>>>(W_ih_in,  D_EMB, W_hh_in,  D_H, D_G4H, Bcat_in);
    pack_b_kernel<<<GRID(D_KCAT * D_G4H), TB, 0, stream>>>(W_ih_out, D_EMB, W_hh_out, D_H, D_G4H, Bcat_out);
    pack_b_kernel<<<GRID(D_H * D_G4H), TB, 0, stream>>>(nullptr, 0, W_ih_p, D_H, D_G4H, Bp_ihp);
    pack_b_kernel<<<GRID(D_H * D_G4H), TB, 0, stream>>>(nullptr, 0, W_hh_p, D_H, D_G4H, Bp_hhp);
    pack_b_kernel<<<GRID(D_H * D_H),   TB, 0, stream>>>(nullptr, 0, W_mp,   D_H, D_H,   Bp_mp);
    pack_b_kernel<<<GRID(D_H * D_P),   TB, 0, stream>>>(nullptr, 0, W_sm,   D_H, D_P,   Bp_sm);

    // GEMM grids: block tile = 128(M) x 64(N)
    const int g_512_4096 = (D_F / 128) * (D_G4H / 64);  // 256 blocks
    const int g_512_1024 = (D_F / 128) * (D_H   / 64);  // 64 blocks
    const int g_128_1024 = (D_B / 128) * (D_P   / 64);  // 16 blocks

    // --- input encoder LSTM: 128 masked steps ---
    for (int t = 0; t < D_L; ++t) {
        gather_x_kernel<<<GRID(D_F * D_EMB), TB, 0, stream>>>(tokens_in, t, emb_bf, Abuf);
        wmma_gemm_kernel<<<g_512_4096, TB, 0, stream>>>(
            Abuf, D_KCAT, Bcat_in, nullptr, gbuf, D_G4H, D_F, D_G4H, D_KCAT);
        lstm_gates_enc_kernel<<<GRID(D_F * D_H), TB, 0, stream>>>(
            gbuf, b_in, lengths_in, t, c_f32, Abuf);
    }
    // --- output encoder LSTM: 128 masked steps, state carries over ---
    for (int t = 0; t < D_L; ++t) {
        gather_x_kernel<<<GRID(D_F * D_EMB), TB, 0, stream>>>(tokens_out, t, emb_bf, Abuf);
        wmma_gemm_kernel<<<g_512_4096, TB, 0, stream>>>(
            Abuf, D_KCAT, Bcat_out, nullptr, gbuf, D_G4H, D_F, D_G4H, D_KCAT);
        lstm_gates_enc_kernel<<<GRID(D_F * D_H), TB, 0, stream>>>(
            gbuf, b_out, lengths_out, t, c_f32, Abuf);
    }

    // --- decoder setup: hdec = final h; pproj = prev_h @ W_ih_p^T + b_p ---
    copy_h_kernel<<<GRID(D_F * D_H), TB, 0, stream>>>(Abuf, hdec);
    wmma_gemm_kernel<<<g_512_4096, TB, 0, stream>>>(
        Abuf + D_EMB, D_KCAT, Bp_ihp, b_p, pproj, D_G4H, D_F, D_G4H, D_H);

    // --- program decoder: 16 steps ---
    for (int t = 0; t < D_PLEN; ++t) {
        wmma_gemm_kernel<<<g_512_4096, TB, 0, stream>>>(
            hdec, D_H, Bp_hhp, nullptr, gbuf, D_G4H, D_F, D_G4H, D_H);
        lstm_gates_dec_kernel<<<GRID(D_F * D_H), TB, 0, stream>>>(
            gbuf, pproj, c_f32, hdec);
        wmma_gemm_kernel<<<g_512_1024, TB, 0, stream>>>(
            hdec, D_H, Bp_mp, nullptr, ubuf, D_H, D_F, D_H, D_H);
        pool_kernel<<<GRID(D_B * D_H), TB, 0, stream>>>(ubuf, b_mp, pooled);
        wmma_gemm_kernel<<<g_128_1024, TB, 0, stream>>>(
            pooled, D_H, Bp_sm, b_sm, out + (size_t)t * D_B * D_P, D_P,
            D_B, D_P, D_H);
    }
    #undef GRID
}
